// VolumeSDFRenderer_42314017800364
// MI455X (gfx1250) — matrix operand order repaired and checked by reading, not compiled, not run on hardware
//
#include <hip/hip_runtime.h>
#include <hip/hip_bf16.h>

typedef _Float16 h8   __attribute__((ext_vector_type(8)));
typedef _Float16 v16h __attribute__((ext_vector_type(16)));
typedef float    v8f  __attribute__((ext_vector_type(8)));
typedef float    f4   __attribute__((ext_vector_type(4)));

#define ALPHA_C   10.0f
#define INV_BETA  20.0f           // 1/0.05
#define FAR_DELTA 1e10f
#define NPTS      128
#define TPAD      132             // f32 row stride (floats): 132*4B = 33*16B -> b128-aligned rows, conflict-free columns
#define WPAD      136             // f16 row stride (halves): 136*2B = 17*16B -> b128-aligned, conflict-free columns

__device__ __forceinline__ float sdf_sigma(float sdf) {
  // psi = Laplace CDF at s = -sdf; sigma = alpha * psi
  float s = -sdf;
  float e = 0.5f * __expf(-fabsf(s) * INV_BETA);
  return ALPHA_C * ((s <= 0.0f) ? e : (1.0f - e));
}

// -------- geometry zero-fill: 201 MB of zeros, nontemporal b128 stores --------
__global__ void zero_geom(f4* __restrict__ g, long n4) {
  long i = (long)blockIdx.x * blockDim.x + threadIdx.x;
  if (i < n4) {
    f4 z = {0.0f, 0.0f, 0.0f, 0.0f};
    __builtin_nontemporal_store(z, g + i);
  }
}

// -------- main renderer: 1 wave == 16 rays, WMMA compositing --------
__global__ __launch_bounds__(32)
void volsdf_render(const float* __restrict__ sdf,
                   const float* __restrict__ color,
                   const float* __restrict__ len,
                   float* __restrict__ out_color) {
  __shared__ float    s_sdf[16 * TPAD];
  __shared__ float    s_len[16 * TPAD];
  __shared__ _Float16 s_w  [16 * WPAD];          // weights, f16, [ray][p]
  __shared__ _Float16 s_col[3 * 16 * WPAD];      // color,   f16, [c][ray][p]

  const int  lane = threadIdx.x;
  const long r0   = (long)blockIdx.x * 16;

  // ---- stage sdf & lengths tiles (contiguous 8KB each, coalesced b128) ----
  {
    const f4* gs = (const f4*)(sdf + r0 * NPTS);
    const f4* gl = (const f4*)(len + r0 * NPTS);
    for (int i = lane; i < 512; i += 32) {       // 16 iters/lane, uniform
      int t = i * 4, ray = t >> 7, p = t & 127;
      f4 v = __builtin_nontemporal_load(gs + i);
      *(f4*)(s_sdf + ray * TPAD + p) = v;
      f4 w = __builtin_nontemporal_load(gl + i);
      *(f4*)(s_len + ray * TPAD + p) = w;
    }
  }

  // ---- stage color as f16, planar [c][ray][p] (contiguous 24KB read) ----
  {
    const f4* gc = (const f4*)(color + r0 * NPTS * 3);
    for (int i = lane; i < 1536; i += 32) {      // 48 iters/lane, uniform
      f4 v = __builtin_nontemporal_load(gc + i);
      int t = i * 4;
#pragma unroll
      for (int u = 0; u < 4; ++u) {
        int tt = t + u;
        int ray = tt / 384, rem = tt - ray * 384;
        int p = rem / 3,    c   = rem - p * 3;
        s_col[c * (16 * WPAD) + ray * WPAD + p] = (_Float16)v[u];
      }
    }
  }
  __syncthreads();

  // ---- weights via exp-of-cumsum (no serial cumprod):
  //      w_p = exp(-S_p) - exp(-(S_p + x_p)),  x_p = sigma_p * delta_p
  const int m    = lane & 15;                    // ray within tile
  const int hsel = lane >> 4;                    // which 64-sample half
  const float* sd = s_sdf + m * TPAD + hsel * 64;
  const float* ln = s_len + m * TPAD + hsel * 64;

  float S = 0.0f;                                // pass 1: per-half total
  for (int j = 0; j < 64; ++j) {
    int p = hsel * 64 + j;
    float delta = (p == 127) ? FAR_DELTA : (ln[j + 1] - ln[j]);
    S += sdf_sigma(sd[j]) * delta;
  }
  float base = __shfl(S, m, 32);                 // total of lower half (lane m)
  if (hsel == 0) base = 0.0f;

  S = base;                                      // pass 2: emit weights
  _Float16* wrow = s_w + m * WPAD + hsel * 64;
  for (int j = 0; j < 64; ++j) {
    int p = hsel * 64 + j;
    float delta = (p == 127) ? FAR_DELTA : (ln[j + 1] - ln[j]);
    float x = sdf_sigma(sd[j]) * delta;
    wrow[j] = (_Float16)(__expf(-S) - __expf(-(S + x)));
    S += x;
  }
  __syncthreads();

  // ---- build A fragments (ISA 16-bit A 16x32 layout):
  //  lanes 0-15 : row M=lane,    K = k0+[0..7]  and k0+16+[0..7]
  //  lanes 16-31: row M=lane-16, K = k0+8+[0..7] and k0+24+[0..7]
  v16h afrag[4];
#pragma unroll
  for (int ch = 0; ch < 4; ++ch) {
    int k0 = ch * 32;
    h8 a0 = *(const h8*)(s_w + m * WPAD + k0 + 8 * hsel);
    h8 a1 = *(const h8*)(s_w + m * WPAD + k0 + 16 + 8 * hsel);
#pragma unroll
    for (int i = 0; i < 8; ++i) { afrag[ch][i] = a0[i]; afrag[ch][8 + i] = a1[i]; }
  }

  // diagonal ownership of the 16x16 f32 D tile:
  //  lanes 0-7 : D[lane,lane]       in acc[lane]
  //  lanes 24-31: D[lane-16,lane-16] in acc[lane-24]
  const bool own  = (lane < 8) || (lane >= 24);
  const int  dvg  = (lane < 8) ? lane : (lane - 24);
  const int  dray = (lane < 8) ? lane : (lane - 16);

#pragma unroll
  for (int c = 0; c < 3; ++c) {
    v8f acc = {0.f, 0.f, 0.f, 0.f, 0.f, 0.f, 0.f, 0.f};
#pragma unroll
    for (int ch = 0; ch < 4; ++ch) {
      int k0 = ch * 32;
      // B 32x16 layout: lanes 0-15 col N=lane, K=k0+[0..7],k0+16+[0..7]; lanes 16-31 K=k0+8+[0..7],k0+24+[0..7]
      const _Float16* bp = s_col + c * (16 * WPAD) + m * WPAD + k0 + 8 * hsel;
      h8 b0 = *(const h8*)(bp);
      h8 b1 = *(const h8*)(bp + 16);
      v16h bfrag;
#pragma unroll
      for (int i = 0; i < 8; ++i) { bfrag[i] = b0[i]; bfrag[8 + i] = b1[i]; }
      acc = __builtin_amdgcn_wmma_f32_16x16x32_f16(
          /*neg_a=*/false, afrag[ch], /*neg_b=*/false, bfrag,
          /*c_mod=*/(short)0, acc, /*reuse_a=*/false, /*reuse_b=*/false);
    }
    float v = acc[0];
    if (dvg == 1) v = acc[1];
    if (dvg == 2) v = acc[2];
    if (dvg == 3) v = acc[3];
    if (dvg == 4) v = acc[4];
    if (dvg == 5) v = acc[5];
    if (dvg == 6) v = acc[6];
    if (dvg == 7) v = acc[7];
    if (own) out_color[(r0 + dray) * 3 + c] = v;
  }
}

extern "C" void kernel_launch(void* const* d_in, const int* in_sizes, int n_in,
                              void* d_out, int out_size, void* d_ws, size_t ws_size,
                              hipStream_t stream) {
  const float* sdf   = (const float*)d_in[0];   // [N,128]
  const float* color = (const float*)d_in[1];   // [N,128,3]
  const float* len   = (const float*)d_in[2];   // [N,128]
  float* out = (float*)d_out;                   // [N,3] ++ [N,128,3]

  const int N = in_sizes[0] / NPTS;
  float* out_color = out;
  f4*    geom      = (f4*)(out + (long)N * 3);
  long   n4        = (long)N * NPTS * 3 / 4;

  hipLaunchKernelGGL(zero_geom, dim3((unsigned)((n4 + 255) / 256)), dim3(256),
                     0, stream, geom, n4);
  hipLaunchKernelGGL(volsdf_render, dim3(N / 16), dim3(32),
                     0, stream, sdf, color, len, out_color);
}
